// VQVAECodebookVanilla_34531537060172
// MI455X (gfx1250) — matrix operand order repaired and checked by reading, not compiled, not run on hardware
//
#include <hip/hip_runtime.h>
#include <cstdint>

typedef __attribute__((ext_vector_type(16))) _Float16 v16h;
typedef __attribute__((ext_vector_type(8)))  float    v8f;

#define K_CODES   1024
#define DIM       64
#define N_ROWS    131072            // 32*64*64 flattened spatial positions
#define ZQ_ELEMS  8388608           // 32*64*64*64
#define ROWS_WG   128
#define THREADS   256
#define KCHUNK    256
#define NCHUNK    (K_CODES / KCHUNK)
#define KTILES    (KCHUNK / 16)     // 16 k-tiles per chunk
#define ROW_DW    36                // f16 codebook row in LDS: 32 dwords data + 4 pad (144B)

// workspace layout (bytes)
#define WS_EMB16   0u               // 1024*64*2  = 131072
#define WS_E2      131072u          // 1024*4
#define WS_COUNTS  135168u          // 1024*4
#define WS_PART    139264u          // 1024*4

struct __align__(16) U4 { uint32_t a, b, c, d; };
struct __align__(16) F4 { float x, y, z, w; };

// ---- cross-lane helpers via ds_swizzle (group-of-32, xor mode) ----
template<int IMM> __device__ __forceinline__ float swzf(float v) {
    return __builtin_bit_cast(float,
        __builtin_amdgcn_ds_swizzle(__builtin_bit_cast(int, v), IMM));
}
template<int IMM> __device__ __forceinline__ int swzi(int v) {
    return __builtin_amdgcn_ds_swizzle(v, IMM);
}
// xor masks: imm = (xor<<10) | and_mask(0x1f)
#define SWZ_X1  0x041F
#define SWZ_X2  0x081F
#define SWZ_X4  0x101F
#define SWZ_X8  0x201F
#define SWZ_X16 0x401F

// load one 32x16 B-operand half (16 f16 = 32B) from a padded LDS codebook row
__device__ __forceinline__ v16h ldsB(const uint32_t* rp, int off) {
    union { U4 u[2]; v16h h; } x;
    x.u[0] = *(const U4*)(rp + off);
    x.u[1] = *(const U4*)(rp + off + 4);
    return x.h;
}

// ============ kernel 1: zero the code-usage histogram ============
__global__ void vq_init(uint8_t* __restrict__ ws) {
    uint32_t* counts = (uint32_t*)(ws + WS_COUNTS);
    counts[blockIdx.x * blockDim.x + threadIdx.x] = 0u;
}

// ============ kernel 2: codebook f32 -> f16 copy + ||e||^2 ============
__global__ void __launch_bounds__(256) vq_prep(const float* __restrict__ emb,
                                               uint8_t* __restrict__ ws) {
    _Float16* e16 = (_Float16*)(ws + WS_EMB16);
    float*    e2  = (float*)(ws + WS_E2);
    const int t = threadIdx.x, lane = t & 31, wave = t >> 5;
    const int r = blockIdx.x * 8 + wave;                 // 128 blocks * 8 waves = 1024 rows
    const float f0 = emb[(size_t)r * 64 + lane];
    const float f1 = emb[(size_t)r * 64 + 32 + lane];
    e16[(size_t)r * 64 + lane]      = (_Float16)f0;
    e16[(size_t)r * 64 + 32 + lane] = (_Float16)f1;
    float s = f0 * f0 + f1 * f1;
    s += swzf<SWZ_X1>(s);  s += swzf<SWZ_X2>(s);  s += swzf<SWZ_X4>(s);
    s += swzf<SWZ_X8>(s);  s += swzf<SWZ_X16>(s);
    if (lane == 0) e2[r] = s;
}

// ============ kernel 3: fused WMMA distances + argmin + all outputs ============
__global__ void __launch_bounds__(THREADS) vq_main(const float* __restrict__ z,
                                                   const float* __restrict__ emb,
                                                   float* __restrict__ out,
                                                   uint8_t* __restrict__ ws) {
    __shared__ uint32_t embS[KCHUNK * ROW_DW];   // 36864 B, padded f16 codebook chunk
    __shared__ float    e2S[KCHUNK];
    __shared__ int      codesS[ROWS_WG];
    __shared__ float    redS[8];

    const float*    e2g    = (const float*)(ws + WS_E2);
    uint32_t*       counts = (uint32_t*)(ws + WS_COUNTS);
    float*          part   = (float*)(ws + WS_PART);
    const U4*       embW   = (const U4*)(ws + WS_EMB16); // 1024 rows * 8 U4
    const uint8_t*  embB   = ws + WS_EMB16;

    const int t    = threadIdx.x;
    const int lane = t & 31;
    const int wave = t >> 5;
    const int kg   = lane >> 4;       // K-group (0/1) in WMMA striping
    const int nCol = lane & 15;       // N column of C/D tile
    const int wgBase = blockIdx.x * ROWS_WG;

    // ---- build A tiles (16 rows x d0..31 and d32..63) in contiguous k order (k = 16*kg + s) ----
    // lane 0-15: halves 0..15 -> K=0..15 ; lane 16-31: K=16..31
    const int R  = wgBase + wave * 16 + nCol;            // row for this lane's M
    const int rb = R >> 12, rh = (R >> 6) & 63, rw = R & 63;
    const float* zb = z + ((size_t)rb * 262144 + (size_t)rh * 64 + (size_t)rw);
    v16h a0, a1;
    #pragma unroll
    for (int i = 0; i < 8; ++i) {
        const int d0 = kg * 16 + i;
        a0[i]     = (_Float16)zb[(size_t)(d0)      * 4096];
        a0[i + 8] = (_Float16)zb[(size_t)(d0 + 8)  * 4096];
        a1[i]     = (_Float16)zb[(size_t)(d0 + 32) * 4096];
        a1[i + 8] = (_Float16)zb[(size_t)(d0 + 40) * 4096];
    }

    // running argmin of (0.5*||e||^2 - x.e) per accumulator slot
    float mv[8]; int mi[8];
    #pragma unroll
    for (int s = 0; s < 8; ++s) { mv[s] = 3.0e38f; mi[s] = 0; }

    const uint32_t* rowp = &embS[nCol * ROW_DW];         // tile kt row: rowp + kt*16*ROW_DW

    for (int c = 0; c < NCHUNK; ++c) {
        __syncthreads();                                  // previous chunk consumed
        #pragma unroll
        for (int i = t; i < KCHUNK * 8; i += THREADS) {   // coalesced b128 fill
            const int r = i >> 3, q = i & 7;
            U4 v = embW[(size_t)(c * KCHUNK + r) * 8 + q];
            *(U4*)&embS[r * ROW_DW + q * 4] = v;
        }
        e2S[t] = e2g[c * KCHUNK + t];
        // pull next chunk's 32 KB toward the WGP while this chunk computes
        if (c + 1 < NCHUNK)
            __builtin_prefetch(embB + (size_t)(c + 1) * (KCHUNK * DIM * 2) + t * 128, 0, 1);
        __syncthreads();

        // Software-pipelined: ds_load_b128 for tiles kt+2/kt+3 issue before the
        // WMMAs of tiles kt/kt+1 execute, so no s_wait_dscnt sits in front of a
        // WMMA. Two accumulator chains hide the WMMA C-chain hazard.
        v16h c00 = ldsB(rowp,                  kg * 8);
        v16h c01 = ldsB(rowp,             16 + kg * 8);
        v16h c10 = ldsB(rowp + 16 * ROW_DW,    kg * 8);
        v16h c11 = ldsB(rowp + 16 * ROW_DW, 16 + kg * 8);
        #pragma unroll
        for (int kt = 0; kt < KTILES; kt += 2) {
            v16h n00, n01, n10, n11;
            if (kt + 2 < KTILES) {
                const uint32_t* rpn = rowp + (kt + 2) * 16 * ROW_DW;
                n00 = ldsB(rpn,                  kg * 8);
                n01 = ldsB(rpn,             16 + kg * 8);
                n10 = ldsB(rpn + 16 * ROW_DW,    kg * 8);
                n11 = ldsB(rpn + 16 * ROW_DW, 16 + kg * 8);
            }
            const float he2a = 0.5f * e2S[kt * 16 + nCol];
            const float he2b = 0.5f * e2S[kt * 16 + 16 + nCol];

            v8f cz = {0.f, 0.f, 0.f, 0.f, 0.f, 0.f, 0.f, 0.f};
            v8f acc0 = __builtin_amdgcn_wmma_f32_16x16x32_f16(
                false, a0, false, c00, (short)0, cz, false, false);
            v8f acc1 = __builtin_amdgcn_wmma_f32_16x16x32_f16(
                false, a0, false, c10, (short)0, cz, false, false);
            acc0 = __builtin_amdgcn_wmma_f32_16x16x32_f16(
                false, a1, false, c01, (short)0, acc0, false, false);
            acc1 = __builtin_amdgcn_wmma_f32_16x16x32_f16(
                false, a1, false, c11, (short)0, acc1, false, false);

            const int kidx0 = (c * KTILES + kt) * 16 + nCol;
            const int kidx1 = kidx0 + 16;
            #pragma unroll
            for (int s = 0; s < 8; ++s) {
                const float sc0 = he2a - acc0[s];
                if (sc0 < mv[s]) { mv[s] = sc0; mi[s] = kidx0; }  // strict < keeps lowest k
            }
            #pragma unroll
            for (int s = 0; s < 8; ++s) {
                const float sc1 = he2b - acc1[s];
                if (sc1 < mv[s]) { mv[s] = sc1; mi[s] = kidx1; }
            }
            c00 = n00; c01 = n01; c10 = n10; c11 = n11;
        }
    }

    // ---- cross-lane argmin within each 16-lane group (ties -> lowest index) ----
    #pragma unroll
    for (int s = 0; s < 8; ++s) {
        float v = mv[s]; int ix = mi[s];
        #define ARGMIN_STEP(IMM) { float ov = swzf<IMM>(v); int oi = swzi<IMM>(ix); \
            const bool tk = (ov < v) | ((ov == v) & (oi < ix)); \
            v = tk ? ov : v; ix = tk ? oi : ix; }
        ARGMIN_STEP(SWZ_X1) ARGMIN_STEP(SWZ_X2) ARGMIN_STEP(SWZ_X4) ARGMIN_STEP(SWZ_X8)
        #undef ARGMIN_STEP
        mv[s] = v; mi[s] = ix;
    }
    if (nCol == 0) {
        #pragma unroll
        for (int s = 0; s < 8; ++s) {
            codesS[wave * 16 + kg * 8 + s] = mi[s];
            atomicAdd(&counts[mi[s]], 1u);               // integer -> deterministic
        }
    }
    __syncthreads();

    // ---- write one-hot encodings (512 MB stream, float4, fully coalesced) ----
    float* enc = out + (size_t)(2 + ZQ_ELEMS);
    for (int i = t; i < ROWS_WG * 256; i += THREADS) {   // 256 float4 per row
        const int r = i >> 8, c4 = (i & 255) * 4;
        const int code = codesS[r];
        F4 vv;
        vv.x = (c4 + 0 == code) ? 1.f : 0.f;
        vv.y = (c4 + 1 == code) ? 1.f : 0.f;
        vv.z = (c4 + 2 == code) ? 1.f : 0.f;
        vv.w = (c4 + 3 == code) ? 1.f : 0.f;
        *(F4*)(enc + (size_t)(wgBase + r) * 1024 + c4) = vv;
    }

    // ---- z_q gather (NCHW, w-coalesced) + loss partial ----
    float lacc = 0.f;
    float* zq = out + 1;
    for (int i = t; i < ROWS_WG * DIM; i += THREADS) {
        const int r = i & (ROWS_WG - 1), d = i >> 7;
        const int Rg = wgBase + r;
        const int gb = Rg >> 12, gh = (Rg >> 6) & 63, gw = Rg & 63;
        const size_t addr = (size_t)gb * 262144 + (size_t)d * 4096 + (size_t)gh * 64 + gw;
        const float ev = emb[(size_t)codesS[r] * 64 + d];   // codebook L2-resident
        const float zv = z[addr];
        zq[addr] = ev;                                      // straight-through == z_q
        const float df = ev - zv;
        lacc = fmaf(df, df, lacc);
    }
    lacc += swzf<SWZ_X1>(lacc); lacc += swzf<SWZ_X2>(lacc); lacc += swzf<SWZ_X4>(lacc);
    lacc += swzf<SWZ_X8>(lacc); lacc += swzf<SWZ_X16>(lacc);
    if (lane == 0) redS[wave] = lacc;
    __syncthreads();
    if (t == 0) {
        float s2 = 0.f;
        #pragma unroll
        for (int i = 0; i < 8; ++i) s2 += redS[i];
        part[blockIdx.x] = s2;                              // fixed order -> deterministic
    }
}

// ============ kernel 4: finalize vq_loss + perplexity ============
__global__ void __launch_bounds__(256) vq_final(float* __restrict__ out,
                                                uint8_t* __restrict__ ws) {
    const float*    part   = (const float*)(ws + WS_PART);
    const uint32_t* counts = (const uint32_t*)(ws + WS_COUNTS);
    __shared__ float rA[8], rB[8];
    const int t = threadIdx.x, lane = t & 31, wave = t >> 5;
    float ls = 0.f, hs = 0.f;
    for (int i = t; i < 1024; i += 256) {
        ls += part[i];
        const float p = (float)counts[i] * (1.0f / (float)N_ROWS);
        hs += p * logf(p + 1e-10f);
    }
    ls += swzf<SWZ_X1>(ls); ls += swzf<SWZ_X2>(ls); ls += swzf<SWZ_X4>(ls);
    ls += swzf<SWZ_X8>(ls); ls += swzf<SWZ_X16>(ls);
    hs += swzf<SWZ_X1>(hs); hs += swzf<SWZ_X2>(hs); hs += swzf<SWZ_X4>(hs);
    hs += swzf<SWZ_X8>(hs); hs += swzf<SWZ_X16>(hs);
    if (lane == 0) { rA[wave] = ls; rB[wave] = hs; }
    __syncthreads();
    if (t == 0) {
        float L = 0.f, H = 0.f;
        #pragma unroll
        for (int i = 0; i < 8; ++i) { L += rA[i]; H += rB[i]; }
        out[0]            = 1.25f * L / (float)ZQ_ELEMS;    // commitment(0.25) + e2z(1.0)
        out[1 + ZQ_ELEMS] = expf(-H);                       // perplexity
    }
}

extern "C" void kernel_launch(void* const* d_in, const int* in_sizes, int n_in,
                              void* d_out, int out_size, void* d_ws, size_t ws_size,
                              hipStream_t stream) {
    (void)in_sizes; (void)n_in; (void)out_size; (void)ws_size;
    const float* z   = (const float*)d_in[0];   // [32,64,64,64] f32 NCHW
    const float* emb = (const float*)d_in[1];   // [1024,64] f32
    float*   out = (float*)d_out;
    uint8_t* ws  = (uint8_t*)d_ws;

    vq_init <<<4,    256, 0, stream>>>(ws);
    vq_prep <<<128,  256, 0, stream>>>(emb, ws);
    vq_main <<<N_ROWS / ROWS_WG, THREADS, 0, stream>>>(z, emb, out, ws);
    vq_final<<<1,    256, 0, stream>>>(out, ws);
}
